// _RefSparseMoeBlock_8916352106883
// MI455X (gfx1250) — compile-verified, matrix operand order checked by reading, not run on hardware
//
#include <hip/hip_runtime.h>
#include <math.h>

#define TOK 4096          // B*S
#define H   2048
#define FFN 1024
#define NE  8
#define KTOT (NE*FFN)     // 8192

typedef __attribute__((ext_vector_type(16))) __bf16 v16bf;
typedef __attribute__((ext_vector_type(8)))  float  v8f;

struct BfPack { uint4 lo, hi; };

#if __has_builtin(__builtin_amdgcn_global_load_async_to_lds_b128)
#define HAS_ASYNC_LDS 1
typedef int v4i_vec __attribute__((vector_size(16)));
typedef __attribute__((address_space(1))) v4i_vec gv4i;
typedef __attribute__((address_space(3))) v4i_vec lv4i;
#else
#define HAS_ASYNC_LDS 0
#endif

// Stage one 16-byte chunk global -> LDS.
static __device__ __forceinline__ void stage16(const unsigned short* g,
                                               unsigned short* l) {
#if HAS_ASYNC_LDS
  __builtin_amdgcn_global_load_async_to_lds_b128((gv4i*)(void*)g,
                                                 (lv4i*)(void*)l, 0, 0);
#else
  *(uint4*)l = *(const uint4*)g;
#endif
}

static __device__ __forceinline__ void async_wait() {
#if HAS_ASYNC_LDS
#if __has_builtin(__builtin_amdgcn_s_wait_asynccnt)
  __builtin_amdgcn_s_wait_asynccnt(0);
#else
  asm volatile("s_wait_asynccnt 0x0" ::: "memory");
#endif
#endif
}

static __device__ __forceinline__ unsigned short f2bf(float f) {
  unsigned u = __builtin_bit_cast(unsigned, f);
  unsigned r = u + 0x7fffu + ((u >> 16) & 1u);   // round-to-nearest-even
  return (unsigned short)(r >> 16);
}

// Build a 16x bf16 WMMA fragment from two 16B LDS chunks (K 0..7 / 16..23 per
// lane-half, matching the CDNA5 16-bit A/B VGPR layout).
static __device__ __forceinline__ v16bf ldfrag(const unsigned short* p) {
  BfPack pk;
  pk.lo = *(const uint4*)(p);
  pk.hi = *(const uint4*)(p + 16);
  return __builtin_bit_cast(v16bf, pk);
}

// ---------------------------------------------------------------- cast f32->bf16
__global__ __launch_bounds__(256) void cast_bf16_kernel(
    const float* __restrict__ src, unsigned short* __restrict__ dst, int n4) {
  int i = blockIdx.x * 256 + threadIdx.x;
  if (i >= n4) return;
  float4 v = ((const float4*)src)[i];
  uint2 p;
  p.x = (unsigned)f2bf(v.x) | ((unsigned)f2bf(v.y) << 16);
  p.y = (unsigned)f2bf(v.z) | ((unsigned)f2bf(v.w) << 16);
  ((uint2*)dst)[i] = p;
}

// ---------------------------------------------------------------- router (tiny)
__global__ __launch_bounds__(256) void router_kernel(
    const float* __restrict__ x, const float* __restrict__ Wg,
    const float* __restrict__ bias, float* __restrict__ expw) {
  int t = blockIdx.x * 256 + threadIdx.x;
  if (t >= TOK) return;
  const float4* xr = (const float4*)(x + (size_t)t * H);
  float acc[NE] = {};
  for (int k = 0; k < H / 4; k++) {
    float4 xv = xr[k];
#pragma unroll
    for (int e = 0; e < NE; e++) {
      float4 wv = ((const float4*)(Wg + (size_t)e * H))[k];
      acc[e] += xv.x * wv.x + xv.y * wv.y + xv.z * wv.z + xv.w * wv.w;
    }
  }
  float sig[NE], sc[NE];
#pragma unroll
  for (int e = 0; e < NE; e++) {
    sig[e] = 1.0f / (1.0f + __expf(-acc[e]));
    sc[e]  = sig[e] + bias[e];
  }
  int i0 = 0;
#pragma unroll
  for (int e = 1; e < NE; e++) if (sc[e] > sc[i0]) i0 = e;     // lowest-index tie-break
  int i1 = (i0 == 0) ? 1 : 0;
#pragma unroll
  for (int e = 0; e < NE; e++) if (e != i0 && sc[e] > sc[i1]) i1 = e;
  float inv = 1.0f / (sig[i0] + sig[i1]);
  float w0 = sig[i0] * inv, w1 = sig[i1] * inv;
#pragma unroll
  for (int e = 0; e < NE; e++)
    expw[(size_t)t * NE + e] = (e == i0) ? w0 : ((e == i1) ? w1 : 0.0f);
}

// -------------------------------------------------- up-proj + SwiGLU + gate (WMMA)
// grid: (FFN/64, TOK/128, NE)   block: 256 (8 waves: 4 in M x 2 in N, wave tile 32x32)
__global__ __launch_bounds__(256) void moe_up_kernel(
    const unsigned short* __restrict__ xb,
    const unsigned short* __restrict__ w1b,
    const unsigned short* __restrict__ w3b,
    const float* __restrict__ expw,
    unsigned short* __restrict__ gbuf) {
  __shared__ unsigned short Xs[2][128 * 32];
  __shared__ unsigned short W1s[2][64 * 32];
  __shared__ unsigned short W3s[2][64 * 32];

  const int e = blockIdx.z;
  const int nBase   = blockIdx.x * 64;
  const int tokBase = blockIdx.y * 128;
  const int tid  = threadIdx.x;
  const int lane = tid & 31;
  const int wid  = tid >> 5;
  const int waveM = wid & 3;   // 0..3  -> 32-row slice
  const int waveN = wid >> 2;  // 0..1  -> 32-col slice

  v8f acc1[2][2] = {};
  v8f acc3[2][2] = {};

  const int r  = lane & 15;
  const int kb = (lane >> 4) << 3;

  // exactly 2 X-chunks + 1 W1-chunk + 1 W3-chunk per thread (no predication)
  const int xr0 = tid >> 2,          xc0 = (tid & 3) << 3;
  const int xr1 = (tid + 256) >> 2,  xc1 = xc0;
  const int wr  = tid >> 2,          wc  = (tid & 3) << 3;

  auto stageUp = [&](int k0, int buf) {
    stage16(&xb[(size_t)(tokBase + xr0) * H + k0 + xc0], &Xs[buf][xr0 * 32 + xc0]);
    stage16(&xb[(size_t)(tokBase + xr1) * H + k0 + xc1], &Xs[buf][xr1 * 32 + xc1]);
    size_t wb = (size_t)e * FFN * H + (size_t)(nBase + wr) * H + k0 + wc;
    stage16(&w1b[wb], &W1s[buf][wr * 32 + wc]);
    stage16(&w3b[wb], &W3s[buf][wr * 32 + wc]);
  };

  stageUp(0, 0);
  async_wait();
  __syncthreads();

  int cur = 0;
  for (int k0 = 0; k0 < H; k0 += 32) {
    if (k0 + 32 < H) stageUp(k0 + 32, cur ^ 1);   // prefetch next tile (async)

    v16bf A[2], B1[2], B3[2];
#pragma unroll
    for (int i = 0; i < 2; i++)
      A[i] = ldfrag(&Xs[cur][(waveM * 32 + i * 16 + r) * 32 + kb]);
#pragma unroll
    for (int j = 0; j < 2; j++) {
      B1[j] = ldfrag(&W1s[cur][(waveN * 32 + j * 16 + r) * 32 + kb]);
      B3[j] = ldfrag(&W3s[cur][(waveN * 32 + j * 16 + r) * 32 + kb]);
    }
#pragma unroll
    for (int i = 0; i < 2; i++)
#pragma unroll
      for (int j = 0; j < 2; j++) {
        acc1[i][j] = __builtin_amdgcn_wmma_f32_16x16x32_bf16(
            false, A[i], false, B1[j], (short)0, acc1[i][j], false, false);
        acc3[i][j] = __builtin_amdgcn_wmma_f32_16x16x32_bf16(
            false, A[i], false, B3[j], (short)0, acc3[i][j], false, false);
      }
    async_wait();
    __syncthreads();
    cur ^= 1;
  }

  // epilogue: g = silu(h1) * h3 * expert_w, store bf16 into [T][E*FFN]
  const int rowHalf = (lane >> 4) << 3;
  const int colLane = lane & 15;
#pragma unroll
  for (int i = 0; i < 2; i++)
#pragma unroll
    for (int j = 0; j < 2; j++) {
      int col = nBase + waveN * 32 + j * 16 + colLane;
#pragma unroll
      for (int r2 = 0; r2 < 8; r2++) {
        int t = tokBase + waveM * 32 + i * 16 + rowHalf + r2;
        float w   = expw[(size_t)t * NE + e];
        float h1v = acc1[i][j][r2];
        float h3v = acc3[i][j][r2];
        float sv  = h1v / (1.0f + __expf(-h1v));
        gbuf[(size_t)t * KTOT + e * FFN + col] = f2bf(sv * h3v * w);
      }
    }
}

// -------------------------------------------------- down-proj: out = g @ W2' (WMMA)
// Single GEMM: M=TOK, N=H, K=E*FFN.  grid: (H/128, TOK/128)  block: 256
// wave tile 32x64 (2x4 accumulators)
__global__ __launch_bounds__(256) void moe_down_kernel(
    const unsigned short* __restrict__ gbuf,
    const unsigned short* __restrict__ w2b,
    float* __restrict__ out) {
  __shared__ unsigned short As[2][128 * 32];
  __shared__ unsigned short Bs[2][128 * 32];
  const int nBase   = blockIdx.x * 128;
  const int tokBase = blockIdx.y * 128;
  const int tid  = threadIdx.x;
  const int lane = tid & 31, wid = tid >> 5;
  const int waveM = wid & 3, waveN = wid >> 2;

  v8f acc[2][4] = {};
  const int r  = lane & 15;
  const int kb = (lane >> 4) << 3;

  const int r0 = tid >> 2,          c0 = (tid & 3) << 3;
  const int r1 = (tid + 256) >> 2,  c1 = c0;

  auto stageDn = [&](int k0, int buf) {
    const int e  = k0 >> 10;          // chunk stays inside one expert (1024 % 32 == 0)
    const int f0 = k0 & (FFN - 1);
    stage16(&gbuf[(size_t)(tokBase + r0) * KTOT + k0 + c0], &As[buf][r0 * 32 + c0]);
    stage16(&gbuf[(size_t)(tokBase + r1) * KTOT + k0 + c1], &As[buf][r1 * 32 + c1]);
    stage16(&w2b[(size_t)e * H * FFN + (size_t)(nBase + r0) * FFN + f0 + c0],
            &Bs[buf][r0 * 32 + c0]);
    stage16(&w2b[(size_t)e * H * FFN + (size_t)(nBase + r1) * FFN + f0 + c1],
            &Bs[buf][r1 * 32 + c1]);
  };

  stageDn(0, 0);
  async_wait();
  __syncthreads();

  int cur = 0;
  for (int k0 = 0; k0 < KTOT; k0 += 32) {
    if (k0 + 32 < KTOT) stageDn(k0 + 32, cur ^ 1);   // prefetch next tile (async)

    v16bf A[2], Bv[4];
#pragma unroll
    for (int i = 0; i < 2; i++)
      A[i] = ldfrag(&As[cur][(waveM * 32 + i * 16 + r) * 32 + kb]);
#pragma unroll
    for (int j = 0; j < 4; j++)
      Bv[j] = ldfrag(&Bs[cur][(waveN * 64 + j * 16 + r) * 32 + kb]);
#pragma unroll
    for (int i = 0; i < 2; i++)
#pragma unroll
      for (int j = 0; j < 4; j++)
        acc[i][j] = __builtin_amdgcn_wmma_f32_16x16x32_bf16(
            false, A[i], false, Bv[j], (short)0, acc[i][j], false, false);
    async_wait();
    __syncthreads();
    cur ^= 1;
  }

  const int rowHalf = (lane >> 4) << 3;
  const int colLane = lane & 15;
#pragma unroll
  for (int i = 0; i < 2; i++)
#pragma unroll
    for (int j = 0; j < 4; j++) {
      int col = nBase + waveN * 64 + j * 16 + colLane;
#pragma unroll
      for (int r2 = 0; r2 < 8; r2++) {
        int t = tokBase + waveM * 32 + i * 16 + rowHalf + r2;
        out[(size_t)t * H + col] = acc[i][j][r2];
      }
    }
}

// ---------------------------------------------------------------- launch
extern "C" void kernel_launch(void* const* d_in, const int* in_sizes, int n_in,
                              void* d_out, int out_size, void* d_ws, size_t ws_size,
                              hipStream_t stream) {
  (void)in_sizes; (void)n_in; (void)out_size; (void)ws_size;
  // setup_inputs order: hidden_states, Wg, W1, W2, W3, bias
  const float* x    = (const float*)d_in[0];
  const float* Wg   = (const float*)d_in[1];
  const float* W1   = (const float*)d_in[2];
  const float* W2   = (const float*)d_in[3];
  const float* W3   = (const float*)d_in[4];
  const float* bias = (const float*)d_in[5];
  float* out = (float*)d_out;

  unsigned short* xb  = (unsigned short*)d_ws;              // T*H bf16
  unsigned short* w1b = xb  + (size_t)TOK * H;              // E*FFN*H bf16
  unsigned short* w3b = w1b + (size_t)NE * FFN * H;
  unsigned short* w2b = w3b + (size_t)NE * FFN * H;         // E*H*FFN bf16
  unsigned short* gb  = w2b + (size_t)NE * H * FFN;         // T*(E*FFN) bf16
  float* expw = (float*)(gb + (size_t)TOK * KTOT);          // T*E f32

  const int n4x = TOK * H / 4;
  const int n4w = NE * FFN * H / 4;
  cast_bf16_kernel<<<(n4x + 255) / 256, 256, 0, stream>>>(x,  xb,  n4x);
  cast_bf16_kernel<<<(n4w + 255) / 256, 256, 0, stream>>>(W1, w1b, n4w);
  cast_bf16_kernel<<<(n4w + 255) / 256, 256, 0, stream>>>(W3, w3b, n4w);
  cast_bf16_kernel<<<(n4w + 255) / 256, 256, 0, stream>>>(W2, w2b, n4w);
  router_kernel<<<TOK / 256, 256, 0, stream>>>(x, Wg, bias, expw);
  moe_up_kernel<<<dim3(FFN / 64, TOK / 128, NE), 256, 0, stream>>>(xb, w1b, w3b, expw, gb);
  moe_down_kernel<<<dim3(H / 128, TOK / 128), 256, 0, stream>>>(gb, w2b, out);
}
